// EdgeClassifierGATStats_6820408066561
// MI455X (gfx1250) — matrix operand order repaired and checked by reading, not compiled, not run on hardware
//
#include <hip/hip_runtime.h>

// ---------------- problem constants ----------------
#define N_NODES 50000
#define N_EDGES 800000
#define E2      (N_EDGES + N_NODES)   // with virtual self loops
#define IN_CH   128
#define HID     64
#define HEADS   4
#define C1      (HEADS * HID)         // 256
#define OUT_CH  2
#define MLP_K   131                   // 2*HID + 3
#define MLP_KP  160                   // padded to multiple of 32
#define NEG_SLOPE 0.2f
#define EPS 1e-16f

typedef __attribute__((ext_vector_type(16))) __bf16 v16bf;
typedef __attribute__((ext_vector_type(8)))  float  v8f;

union FragU { uint4 q[2]; v16bf v; };

// ---------------- helpers ----------------
__device__ __forceinline__ unsigned short f2bf(float f) {
    unsigned u = __float_as_uint(f);
    unsigned r = u + 0x7FFFu + ((u >> 16) & 1u);   // round-nearest-even
    return (unsigned short)(r >> 16);
}
// order-preserving float<->uint encoding for atomicMax on floats
__device__ __forceinline__ unsigned enc_f(float f) {
    unsigned u = __float_as_uint(f);
    return (u & 0x80000000u) ? ~u : (u | 0x80000000u);
}
__device__ __forceinline__ float dec_f(unsigned u) {
    unsigned b = (u & 0x80000000u) ? (u ^ 0x80000000u) : ~u;
    return __uint_as_float(b);
}

// ---------------- conversion kernels ----------------
__global__ void k_cvt_bf16(const float* __restrict__ in,
                           unsigned short* __restrict__ out, int n) {
    int i = blockIdx.x * blockDim.x + threadIdx.x;
    if (i < n) out[i] = f2bf(in[i]);
}

// Pre-swizzle a row-major f32 [Kreal x N] weight into the CDNA5 WMMA bf16
// B-fragment order: out[(((kt*ntiles)+nt)*32 + lane)*16 + e] = W[k*N+n]
// with k = kt*32 + 8*(lane>>4) + (e&7) + 16*(e>>3), n = nt*16 + (lane&15). K rows >= Kreal pad 0.
__global__ void k_cvt_wswz(const float* __restrict__ W,
                           unsigned short* __restrict__ out,
                           int Kreal, int Kpad, int N) {
    int idx = blockIdx.x * blockDim.x + threadIdx.x;
    int total = Kpad * N;
    if (idx >= total) return;
    int e    = idx & 15;
    int lane = (idx >> 4) & 31;
    int tile = idx >> 9;
    int ntiles = N >> 4;
    int nt = tile % ntiles;
    int kt = tile / ntiles;
    int k = kt * 32 + ((lane >> 4) << 3) + (e & 7) + ((e >> 3) << 4);
    int n = nt * 16 + (lane & 15);
    float v = (k < Kreal) ? W[(size_t)k * N + n] : 0.0f;
    out[idx] = f2bf(v);
}

// ---------------- bf16 WMMA GEMM: C[M,N] f32 = A[M,K]bf16 @ Bswz ----------------
// one 16x16 tile per wave32; A row-major bf16 (lda=K), B pre-swizzled.
__global__ void k_gemm_bf16(const unsigned short* __restrict__ A,
                            const unsigned short* __restrict__ Bp,
                            float* __restrict__ Cm, int M, int N, int K) {
    int wave  = threadIdx.x >> 5;
    int lane  = threadIdx.x & 31;
    int ntiles = N >> 4;
    int tiles  = (M >> 4) * ntiles;
    int gtile = blockIdx.x * (blockDim.x >> 5) + wave;
    if (gtile >= tiles) return;
    int mt = gtile / ntiles;
    int nt = gtile % ntiles;
    int h = lane >> 4;       // half-wave
    int r = lane & 15;

    const unsigned short* arow = A + (size_t)(mt * 16 + r) * K;
    v8f acc = {};
    for (int kt = 0; kt < (K >> 5); ++kt) {
        FragU fa, fb;
        const unsigned short* ap = arow + kt * 32 + 8 * h;
        fa.q[0] = *(const uint4*)(ap);        // e=0..7  -> K = 8h+e
        fa.q[1] = *(const uint4*)(ap + 16);   // e=8..15 -> K = 16+8h+(e-8)
        const unsigned short* bp = Bp + ((size_t)(kt * ntiles + nt) * 32 + lane) * 16;
        fb.q[0] = *(const uint4*)(bp);
        fb.q[1] = *(const uint4*)(bp + 8);
        acc = __builtin_amdgcn_wmma_f32_16x16x32_bf16(
            false, fa.v, false, fb.v, (short)0, acc, false, false);
    }
#pragma unroll
    for (int v2 = 0; v2 < 8; ++v2)
        Cm[(size_t)(mt * 16 + v2 + 8 * h) * N + nt * 16 + r] = acc[v2];
}

// ---------------- attention coefficients: a_s/a_d per (node, head) -------------
template <int H>
__global__ void k_attc(const float* __restrict__ hf,
                       const float* __restrict__ atts,
                       const float* __restrict__ attd,
                       float* __restrict__ as_, float* __restrict__ ad_) {
    int idx = blockIdx.x * blockDim.x + threadIdx.x;
    if (idx >= N_NODES * H) return;
    int n  = idx / H;
    int hh = idx % H;
    const float* p = hf + ((size_t)n * H + hh) * 64;
    float ss = 0.f, dd = 0.f;
#pragma unroll 8
    for (int c = 0; c < 64; ++c) {
        float v = p[c];
        ss += v * atts[hh * 64 + c];
        dd += v * attd[hh * 64 + c];
    }
    as_[idx] = ss;
    ad_[idx] = dd;
}

// ---------------- edge softmax passes ----------------
template <int H>
__global__ void k_emax(const long long* __restrict__ ei,
                       const float* __restrict__ as_,
                       const float* __restrict__ ad_,
                       unsigned* __restrict__ menc) {
    int i = blockIdx.x * blockDim.x + threadIdx.x;
    if (i >= E2) return;
    int s, d;
    if (i < N_EDGES) { s = (int)ei[i]; d = (int)ei[N_EDGES + i]; }
    else             { s = d = i - N_EDGES; }
#pragma unroll
    for (int hh = 0; hh < H; ++hh) {
        float e = as_[s * H + hh] + ad_[d * H + hh];
        e = (e > 0.f) ? e : NEG_SLOPE * e;
        atomicMax(&menc[d * H + hh], enc_f(e));
    }
}

template <int H>
__global__ void k_esum(const long long* __restrict__ ei,
                       const float* __restrict__ as_,
                       const float* __restrict__ ad_,
                       const unsigned* __restrict__ menc,
                       float* __restrict__ ee, float* __restrict__ ssum) {
    int i = blockIdx.x * blockDim.x + threadIdx.x;
    if (i >= E2) return;
    int s, d;
    if (i < N_EDGES) { s = (int)ei[i]; d = (int)ei[N_EDGES + i]; }
    else             { s = d = i - N_EDGES; }
#pragma unroll
    for (int hh = 0; hh < H; ++hh) {
        float e = as_[s * H + hh] + ad_[d * H + hh];
        e = (e > 0.f) ? e : NEG_SLOPE * e;
        float m  = dec_f(menc[d * H + hh]);
        float ex = expf(e - m);
        ee[(size_t)i * H + hh] = ex;
        atomicAdd(&ssum[d * H + hh], ex);
    }
}

// scatter: out[dst] += alpha * h[src]; one wave per edge, lanes stride channels
template <int H>
__global__ void k_escat(const long long* __restrict__ ei,
                        const float* __restrict__ ee,
                        const float* __restrict__ ssum,
                        const float* __restrict__ hfeat,
                        float* __restrict__ outp) {
    int wave = threadIdx.x >> 5;
    int lane = threadIdx.x & 31;
    int i = blockIdx.x * (blockDim.x >> 5) + wave;
    if (i >= E2) return;
    int s, d;
    if (i < N_EDGES) { s = (int)ei[i]; d = (int)ei[N_EDGES + i]; }
    else             { s = d = i - N_EDGES; }
    float alpha[H];
#pragma unroll
    for (int hh = 0; hh < H; ++hh)
        alpha[hh] = ee[(size_t)i * H + hh] / (ssum[d * H + hh] + EPS);
    const int CH = H * 64;
    const float* hs = hfeat + (size_t)s * CH;
    float* od = outp + (size_t)d * CH;
#pragma unroll
    for (int ch = lane; ch < CH; ch += 32) {
        int hh = ch >> 6;
        atomicAdd(&od[ch], alpha[hh] * hs[ch]);
    }
}

// ---------------- epilogues ----------------
__global__ void k_elub(float* __restrict__ out1, const float* __restrict__ b1,
                       unsigned short* __restrict__ h1b) {
    int i = blockIdx.x * blockDim.x + threadIdx.x;
    if (i >= N_NODES * C1) return;
    float v = out1[i] + b1[i & (C1 - 1)];
    float r = (v > 0.f) ? v : (expf(v) - 1.0f);   // ELU
    out1[i] = r;
    h1b[i] = f2bf(r);
}

__global__ void k_embf(const float* __restrict__ out2, const float* __restrict__ b2,
                       unsigned short* __restrict__ embb) {
    int i = blockIdx.x * blockDim.x + threadIdx.x;
    if (i >= N_NODES * HID) return;
    embb[i] = f2bf(out2[i] + b2[i & (HID - 1)]);
}

// ---------------- edge MLP: 16 edges per block, WMMA over K=160 ----------------
__global__ void __launch_bounds__(128)
k_mlp(const unsigned short* __restrict__ embb,
      const long long* __restrict__ ei,
      const float* __restrict__ stats,
      const unsigned short* __restrict__ mw1b,   // swizzled [160 x 64] bf16
      const float* __restrict__ mb1,
      const float* __restrict__ mW2,             // [64 x 2] f32
      const float* __restrict__ mb2,
      float* __restrict__ out) {
    __shared__ unsigned short As[16 * MLP_KP];   // 16 edges x 160 feats (bf16)
    __shared__ float Hs[16 * HID];               // hidden activations
    __shared__ long long Ssrc[16], Sdst[16];

    const int tid = threadIdx.x;
    const int e0 = blockIdx.x * 16;

    if (tid < 16) {
        long long s_ = ei[e0 + tid];
        long long d_ = ei[N_EDGES + e0 + tid];
        Ssrc[tid] = s_;
        Sdst[tid] = d_;
        __builtin_prefetch(embb + s_ * HID);     // -> global_prefetch_b8
        __builtin_prefetch(embb + d_ * HID);
    }
    __syncthreads();

    // build A tile: [emb[src] | emb[dst] | stats | zero-pad]
    for (int idx = tid; idx < 16 * MLP_KP; idx += 128) {
        int r = idx / MLP_KP;
        int f = idx - r * MLP_KP;
        unsigned short v = 0;
        if (f < 64)        v = embb[Ssrc[r] * HID + f];
        else if (f < 128)  v = embb[Sdst[r] * HID + (f - 64)];
        else if (f < MLP_K) v = f2bf(stats[(size_t)(e0 + r) * 3 + (f - 128)]);
        As[idx] = v;
    }
    __syncthreads();

    // hidden = relu(A @ mW1 + mb1): 4 waves, one 16x16 N-tile each
    const int wave = tid >> 5;
    const int lane = tid & 31;
    const int h = lane >> 4;
    const int r = lane & 15;
    const int nt = wave;                         // N tile 0..3 (HID=64)
    v8f acc = {};
#pragma unroll
    for (int kt = 0; kt < (MLP_KP >> 5); ++kt) {
        FragU fa, fb;
        const unsigned short* ap = &As[r * MLP_KP + kt * 32 + 8 * h];
        fa.q[0] = *(const uint4*)(ap);
        fa.q[1] = *(const uint4*)(ap + 16);
        const unsigned short* bp = mw1b + ((size_t)(kt * 4 + nt) * 32 + lane) * 16;
        fb.q[0] = *(const uint4*)(bp);
        fb.q[1] = *(const uint4*)(bp + 8);
        acc = __builtin_amdgcn_wmma_f32_16x16x32_bf16(
            false, fa.v, false, fb.v, (short)0, acc, false, false);
    }
    const int ch = nt * 16 + r;
    const float bias = mb1[ch];
#pragma unroll
    for (int v2 = 0; v2 < 8; ++v2) {
        int row = v2 + 8 * h;
        Hs[row * HID + ch] = fmaxf(0.0f, acc[v2] + bias);
    }
    __syncthreads();

    // tiny second layer: hidden[16x64] @ mW2[64x2] + mb2
    if (tid < 32) {
        int el = tid >> 1, o = tid & 1;
        float sum = mb2[o];
#pragma unroll 8
        for (int c = 0; c < HID; ++c) sum += Hs[el * HID + c] * mW2[c * 2 + o];
        out[(size_t)(e0 + el) * 2 + o] = sum;
    }
}

// ---------------- launch ----------------
extern "C" void kernel_launch(void* const* d_in, const int* in_sizes, int n_in,
                              void* d_out, int out_size, void* d_ws, size_t ws_size,
                              hipStream_t stream) {
    const float*     x        = (const float*)d_in[0];
    const long long* ei       = (const long long*)d_in[1];
    const float*     stats    = (const float*)d_in[2];
    const float*     W1       = (const float*)d_in[3];
    const float*     att_src1 = (const float*)d_in[4];
    const float*     att_dst1 = (const float*)d_in[5];
    const float*     b1       = (const float*)d_in[6];
    const float*     W2       = (const float*)d_in[7];
    const float*     att_src2 = (const float*)d_in[8];
    const float*     att_dst2 = (const float*)d_in[9];
    const float*     b2       = (const float*)d_in[10];
    const float*     mW1      = (const float*)d_in[11];
    const float*     mb1      = (const float*)d_in[12];
    const float*     mW2      = (const float*)d_in[13];
    const float*     mb2      = (const float*)d_in[14];
    float* out = (float*)d_out;

    // ---- workspace bump allocator (256B aligned) ----
    size_t off = 0;
    char* base = (char*)d_ws;
    auto alloc = [&](size_t bytes) -> char* {
        char* p = base + off;
        off = (off + bytes + 255) & ~(size_t)255;
        return p;
    };
    unsigned short* xb    = (unsigned short*)alloc((size_t)N_NODES * IN_CH * 2);
    unsigned short* w1b   = (unsigned short*)alloc((size_t)IN_CH * C1 * 2);
    unsigned short* w2b   = (unsigned short*)alloc((size_t)C1 * HID * 2);
    unsigned short* mw1b  = (unsigned short*)alloc((size_t)MLP_KP * HID * 2);
    float*          h1pre = (float*)alloc((size_t)N_NODES * C1 * 4);
    float*          out1  = (float*)alloc((size_t)N_NODES * C1 * 4);
    unsigned short* h1b   = (unsigned short*)alloc((size_t)N_NODES * C1 * 2);
    float*          as1   = (float*)alloc((size_t)N_NODES * HEADS * 4);
    float*          ad1   = (float*)alloc((size_t)N_NODES * HEADS * 4);
    unsigned*       m1    = (unsigned*)alloc((size_t)N_NODES * HEADS * 4);
    float*          s1    = (float*)alloc((size_t)N_NODES * HEADS * 4);
    float*          ee1   = (float*)alloc((size_t)E2 * HEADS * 4);
    float*          h2pre = (float*)alloc((size_t)N_NODES * HID * 4);
    float*          out2  = (float*)alloc((size_t)N_NODES * HID * 4);
    float*          as2   = (float*)alloc((size_t)N_NODES * 4);
    float*          ad2   = (float*)alloc((size_t)N_NODES * 4);
    unsigned*       m2    = (unsigned*)alloc((size_t)N_NODES * 4);
    float*          s2    = (float*)alloc((size_t)N_NODES * 4);
    float*          ee2   = (float*)alloc((size_t)E2 * 4);
    unsigned short* embb  = (unsigned short*)alloc((size_t)N_NODES * HID * 2);
    (void)ws_size; (void)in_sizes; (void)n_in; (void)out_size;

    const int T = 256;
    auto cdiv = [](int a, int b) { return (a + b - 1) / b; };

    // convert inputs / pre-swizzle weights
    k_cvt_bf16<<<cdiv(N_NODES * IN_CH, T), T, 0, stream>>>(x, xb, N_NODES * IN_CH);
    k_cvt_wswz<<<cdiv(IN_CH * C1, T), T, 0, stream>>>(W1, w1b, IN_CH, IN_CH, C1);
    k_cvt_wswz<<<cdiv(C1 * HID, T), T, 0, stream>>>(W2, w2b, C1, C1, HID);
    k_cvt_wswz<<<cdiv(MLP_KP * HID, T), T, 0, stream>>>(mW1, mw1b, MLP_K, MLP_KP, HID);

    // ---- layer 1: h1pre = x @ W1 ----
    {
        int tiles = (N_NODES / 16) * (C1 / 16);
        k_gemm_bf16<<<cdiv(tiles, 8), T, 0, stream>>>(xb, w1b, h1pre, N_NODES, C1, IN_CH);
    }
    k_attc<HEADS><<<cdiv(N_NODES * HEADS, T), T, 0, stream>>>(h1pre, att_src1, att_dst1, as1, ad1);
    hipMemsetAsync(m1, 0, (size_t)N_NODES * HEADS * 4, stream);
    hipMemsetAsync(s1, 0, (size_t)N_NODES * HEADS * 4, stream);
    hipMemsetAsync(out1, 0, (size_t)N_NODES * C1 * 4, stream);
    k_emax<HEADS><<<cdiv(E2, T), T, 0, stream>>>(ei, as1, ad1, m1);
    k_esum<HEADS><<<cdiv(E2, T), T, 0, stream>>>(ei, as1, ad1, m1, ee1, s1);
    k_escat<HEADS><<<cdiv(E2, 8), T, 0, stream>>>(ei, ee1, s1, h1pre, out1);
    k_elub<<<cdiv(N_NODES * C1, T), T, 0, stream>>>(out1, b1, h1b);

    // ---- layer 2: h2pre = h1act @ W2 ----
    {
        int tiles = (N_NODES / 16) * (HID / 16);
        k_gemm_bf16<<<cdiv(tiles, 8), T, 0, stream>>>(h1b, w2b, h2pre, N_NODES, HID, C1);
    }
    k_attc<1><<<cdiv(N_NODES, T), T, 0, stream>>>(h2pre, att_src2, att_dst2, as2, ad2);
    hipMemsetAsync(m2, 0, (size_t)N_NODES * 4, stream);
    hipMemsetAsync(s2, 0, (size_t)N_NODES * 4, stream);
    hipMemsetAsync(out2, 0, (size_t)N_NODES * HID * 4, stream);
    k_emax<1><<<cdiv(E2, T), T, 0, stream>>>(ei, as2, ad2, m2);
    k_esum<1><<<cdiv(E2, T), T, 0, stream>>>(ei, as2, ad2, m2, ee2, s2);
    k_escat<1><<<cdiv(E2, 8), T, 0, stream>>>(ei, ee2, s2, h2pre, out2);
    k_embf<<<cdiv(N_NODES * HID, T), T, 0, stream>>>(out2, b2, embb);

    // ---- edge MLP ----
    k_mlp<<<N_EDGES / 16, 128, 0, stream>>>(embb, ei, stats, mw1b, mb1, mW2, mb2, out);
}
